// RIMCell_32693291057934
// MI455X (gfx1250) — matrix operand order repaired and checked by reading, not compiled, not run on hardware
//
#include <hip/hip_runtime.h>
#include <hip/hip_bf16.h>

typedef __bf16 bf16;
typedef __attribute__((ext_vector_type(16))) __bf16 v16bf;
typedef __attribute__((ext_vector_type(8)))  float   v8f;
typedef unsigned int       u32;
typedef unsigned long long u64;

#define B_    2048
#define IN_   512
#define HID_  256
#define NM_   16
#define VAL_  400
#define VALP_ 416   // VAL padded to K-step multiple
#define G3_   768   // 3*HID

// Use gfx1250 async global->LDS copies (ASYNCcnt) for bf16 weight staging.
#define ASYNC_STAGE 1

// ---------------- gfx1250 async copy helpers --------------------------------

static __device__ __forceinline__ void cp16(void* lds, const void* g) {
#if ASYNC_STAGE
  asm volatile("global_load_async_to_lds_b128 %0, %1, off"
               :
               : "v"((u32)(uintptr_t)lds), "v"((u64)(uintptr_t)g)
               : "memory");
#else
  *(uint4*)lds = *(const uint4*)g;
#endif
}
static __device__ __forceinline__ void cp16_wait() {
#if ASYNC_STAGE
  asm volatile("s_wait_asynccnt 0x0" ::: "memory");
#endif
}

// ---------------- WMMA helpers (CDNA5 16x16x32 bf16, wave32) ----------------

static __device__ __forceinline__ v8f wmma_bf16(v16bf a, v16bf b, v8f c) {
  return __builtin_amdgcn_wmma_f32_16x16x32_bf16(false, a, false, b, (short)0, c,
                                                 false, false);
}

// A fragment 16x32 (MxK): lane l: m=l&15, hi=l>>4; K = hi*16 + 0..15
static __device__ __forceinline__ v16bf frag_a(const bf16* base, int ld) {
  const int lane = threadIdx.x & 31;
  const int m = lane & 15, hi = lane >> 4;
  const bf16* p0 = base + m * ld + hi * 16;
  v16bf f;
#pragma unroll
  for (int i = 0; i < 16; ++i) f[i] = p0[i];
  return f;
}

// B fragment from transposed tile Bt[n][k]: lane l: n=l&15, k=(l>>4)*16 + 0..15
static __device__ __forceinline__ v16bf frag_b(const bf16* base, int ld) {
  const int lane = threadIdx.x & 31;
  const int n = lane & 15, hi = lane >> 4;
  const bf16* p = base + n * ld + hi * 16;
  v16bf f;
#pragma unroll
  for (int i = 0; i < 16; ++i) f[i] = p[i];
  return f;
}

// ---------------- Weight pre-pass: f32 [z][K][N] -> bf16 [z][N][Kp] ----------

__global__ __launch_bounds__(256)
void transpose_bf16_kernel(const float* __restrict__ src, bf16* __restrict__ dst,
                           int K, int N, int Kp) {
  __shared__ float ts[32][33];
  const int z = blockIdx.x;
  const int n0 = blockIdx.y * 32;
  const int k0 = blockIdx.z * 32;
  const int t = threadIdx.x;
  const float* S = src + (long)z * K * N;
  bf16* D = dst + (long)z * N * Kp;
#pragma unroll
  for (int j = 0; j < 4; ++j) {
    const int flat = j * 256 + t;
    const int kk = flat >> 5, nn = flat & 31;
    float v = 0.f;
    if (k0 + kk < K && n0 + nn < N) v = S[(long)(k0 + kk) * N + n0 + nn];
    ts[kk][nn] = v;
  }
  __syncthreads();
#pragma unroll
  for (int j = 0; j < 4; ++j) {
    const int flat = j * 256 + t;
    const int nn = flat >> 5, kk = flat & 31;
    if (n0 + nn < N) D[(long)(n0 + nn) * Kp + k0 + kk] = (bf16)ts[kk][nn];
  }
}

// ---------------- bias2[n,h] = value_b @ x2h_w[n] ---------------------------

__global__ void bias2_kernel(const float* __restrict__ value_b,
                             const float* __restrict__ x2h,
                             float* __restrict__ bias2) {
  const int n = blockIdx.x;
  const int h = blockIdx.y * 256 + threadIdx.x;
  const float* W = x2h + (long)n * VAL_ * G3_;
  float s = 0.f;
  for (int v = 0; v < VAL_; ++v) s += value_b[v] * W[(long)v * G3_ + h];
  bias2[n * G3_ + h] = s;
}

// ---------------- Gating: scores, top-8 mask, 2-way softmax -----------------

__global__ __launch_bounds__(32)
void gate_kernel(const float* __restrict__ option, const float* __restrict__ p_w,
                 const float* __restrict__ p_b, float* __restrict__ maskp0,
                 float* __restrict__ maskv) {
  const long b = blockIdx.x;
  const int lane = threadIdx.x;
  const int nn = lane & 15;
  float s0 = 0.f, s1 = 0.f;
#pragma unroll
  for (int o = 0; o < 8; ++o) {
    const float pw = p_w[o * 16 + nn];
    s0 += option[(b * 2 + 0) * 8 + o] * pw;
    s1 += option[(b * 2 + 1) * 8 + o] * pw;
  }
  s0 += p_b[nn];
  int rank = 0;
  for (int j = 0; j < 16; ++j) {
    const float sj = __shfl(s0, j, 16);
    rank += (sj > s0) || (sj == s0 && j < nn);
  }
  const float mk = (rank < 8) ? 1.f : 0.f;
  const float mx = fmaxf(s0, s1);
  const float e0 = __expf(s0 - mx), e1 = __expf(s1 - mx);
  const float p0 = e0 / (e0 + e1);
  if (lane < 16) {
    maskp0[b * 16 + nn] = mk * p0;
    maskv[b * 16 + nn]  = mk;
  }
}

// ---------------- Generic tiled bf16-WMMA GEMM ------------------------------
// C[z][MxN] = A[z][MxK](f32) * Wt[z][NxKp](bf16, pre-transposed). 8 waves, 32x64.

template <bool OUT_BF16>
__global__ __launch_bounds__(256)
void gemm_kernel(const float* __restrict__ A, long a_rs, long a_zs,
                 const bf16* __restrict__ Wt, long w_zs, int ldw,
                 void* __restrict__ Cv, long c_rs, long c_zs, int N, int K, int Kp) {
  __shared__ __align__(16) bf16 As[32][40];
  __shared__ __align__(16) bf16 Bs[64][40];
  const int t = threadIdx.x;
  const int w = t >> 5;
  const int mrow = w >> 2, ncol = w & 3;
  const int z = blockIdx.z;
  const long m0 = (long)blockIdx.x * 32;
  const int n0 = blockIdx.y * 64;
  const float* Az = A + (long)z * a_zs;
  const bf16* Wz = Wt + (long)z * w_zs;
  const int bc = t >> 2, bch = (t & 3) * 8;   // one 16B chunk per thread
  const int bvalid = (n0 + bc) < N;
  v8f acc = {};
  for (int k0 = 0; k0 < Kp; k0 += 32) {
#pragma unroll
    for (int j = 0; j < 4; ++j) {            // stage A 32x32 (f32 -> bf16)
      const int flat = j * 256 + t;
      const int r = flat >> 5, kk = flat & 31;
      const float val = (k0 + kk < K) ? Az[(m0 + r) * a_rs + k0 + kk] : 0.f;
      As[r][kk] = (bf16)val;
    }
    // stage Wt -> Bt (64 x 32) : async b128 copies
    if (bvalid) {
      cp16(&Bs[bc][bch], Wz + (long)(n0 + bc) * ldw + k0 + bch);
    } else {
      uint4 z4 = {0u, 0u, 0u, 0u};
      *(uint4*)&Bs[bc][bch] = z4;
    }
    cp16_wait();
    __syncthreads();
    const v16bf a = frag_a(&As[mrow * 16][0], 40);
    const v16bf b = frag_b(&Bs[ncol * 16][0], 40);
    acc = wmma_bf16(a, b, acc);
    __syncthreads();
  }
  const int lane = t & 31, nl = lane & 15, hi = lane >> 4;
  const int col = n0 + ncol * 16 + nl;
  if (col < N) {
#pragma unroll
    for (int i = 0; i < 8; ++i) {
      const long row = m0 + mrow * 16 + i + hi * 8;
      const long idx = (long)z * c_zs + row * c_rs + col;
      if (OUT_BF16) ((bf16*)Cv)[idx] = (bf16)acc[i];
      else          ((float*)Cv)[idx] = acc[i];
    }
  }
}

// ---------------- Fused gx/gh GEMMs + GRU -----------------------------------
// One block per (16-batch tile, n). gx = maskp0*(xv @ x2h[n]) + mask*bias2,
// gh = hs[:,n,:] @ h2h[n]; GRU epilogue -> hb.  LDS ~157KB of 320KB WGP.

__global__ __launch_bounds__(256)
void g_gru_kernel(const float* __restrict__ xv, const float* __restrict__ hs,
                  const bf16* __restrict__ x2ht, const bf16* __restrict__ h2ht,
                  const float* __restrict__ bias2, const float* __restrict__ maskp0,
                  const float* __restrict__ maskv, float* __restrict__ hb) {
  __shared__ __align__(16) bf16 As[16][40];
  __shared__ __align__(16) bf16 Bs[G3_][40];
  __shared__ float gxs[16][G3_];
  __shared__ float ghs[16][G3_];
  __shared__ float mp0s[16], ms[16];
  const int t = threadIdx.x;
  const int w = t >> 5, lane = t & 31, nl = lane & 15, hi = lane >> 4;
  const int n = blockIdx.y;
  const long b0 = (long)blockIdx.x * 16;
  if (t < 16) {
    mp0s[t] = maskp0[(b0 + t) * NM_ + n];
    ms[t]   = maskv[(b0 + t) * NM_ + n];
  }
  v8f acc[6];
#pragma unroll
  for (int q = 0; q < 6; ++q) acc[q] = (v8f){};

  // ---- loop 1: XG = xv(16x400) @ x2h_t[n](768x416), K padded to 416 ----
  const bf16* W1 = x2ht + (long)n * G3_ * VALP_;
  for (int k0 = 0; k0 < VALP_; k0 += 32) {
#pragma unroll
    for (int j = 0; j < 2; ++j) {            // A: f32 -> bf16
      const int flat = j * 256 + t;
      const int r = flat >> 5, kk = flat & 31;
      const float val = (k0 + kk < VAL_) ? xv[(b0 + r) * VAL_ + k0 + kk] : 0.f;
      As[r][kk] = (bf16)val;
    }
#pragma unroll
    for (int j = 0; j < 12; ++j) {           // Bt 768x32, async b128
      const int flat = j * 256 + t;
      const int c = flat >> 2, ch = (flat & 3) * 8;
      cp16(&Bs[c][ch], W1 + (long)c * VALP_ + k0 + ch);
    }
    cp16_wait();
    __syncthreads();
    const v16bf a = frag_a(&As[0][0], 40);
#pragma unroll
    for (int q = 0; q < 6; ++q) {
      const v16bf b = frag_b(&Bs[w * 96 + q * 16][0], 40);
      acc[q] = wmma_bf16(a, b, acc[q]);
    }
    __syncthreads();
  }
#pragma unroll
  for (int q = 0; q < 6; ++q) {              // epilogue 1 -> gxs
    const int col = w * 96 + q * 16 + nl;
#pragma unroll
    for (int i = 0; i < 8; ++i) {
      const int row = i + hi * 8;
      gxs[row][col] = mp0s[row] * acc[q][i] + ms[row] * bias2[n * G3_ + col];
      acc[q][i] = 0.f;
    }
  }

  // ---- loop 2: GH = hs[:,n,:](16x256) @ h2h_t[n](768x256) ----
  const bf16* W2 = h2ht + (long)n * G3_ * HID_;
  for (int k0 = 0; k0 < HID_; k0 += 32) {
#pragma unroll
    for (int j = 0; j < 2; ++j) {
      const int flat = j * 256 + t;
      const int r = flat >> 5, kk = flat & 31;
      As[r][kk] = (bf16)hs[(b0 + r) * (NM_ * HID_) + (long)n * HID_ + k0 + kk];
    }
#pragma unroll
    for (int j = 0; j < 12; ++j) {
      const int flat = j * 256 + t;
      const int c = flat >> 2, ch = (flat & 3) * 8;
      cp16(&Bs[c][ch], W2 + (long)c * HID_ + k0 + ch);
    }
    cp16_wait();
    __syncthreads();
    const v16bf a = frag_a(&As[0][0], 40);
#pragma unroll
    for (int q = 0; q < 6; ++q) {
      const v16bf b = frag_b(&Bs[w * 96 + q * 16][0], 40);
      acc[q] = wmma_bf16(a, b, acc[q]);
    }
    __syncthreads();
  }
#pragma unroll
  for (int q = 0; q < 6; ++q) {              // epilogue 2 -> ghs
    const int col = w * 96 + q * 16 + nl;
#pragma unroll
    for (int i = 0; i < 8; ++i) ghs[i + hi * 8][col] = acc[q][i];
  }
  __syncthreads();

  // ---- GRU (needs i_n and h_n separately) ----
#pragma unroll
  for (int j = 0; j < 16; ++j) {
    const int flat = j * 256 + t;
    const int r = flat >> 8, h = flat & 255;
    const float rr = 1.f / (1.f + __expf(-(gxs[r][h] + ghs[r][h])));
    const float zz = 1.f / (1.f + __expf(-(gxs[r][h + 256] + ghs[r][h + 256])));
    const float ng = tanhf(gxs[r][h + 512] + rr * ghs[r][h + 512]);
    const long gi = (b0 + r) * (NM_ * HID_) + (long)n * HID_ + h;
    const float hv = hs[gi];
    hb[gi] = ng + zz * (hv - ng);
  }
}

// ---------------- Attention scores: one WMMA per (b, head) ------------------

__global__ __launch_bounds__(128)
void att_kernel(const bf16* __restrict__ q, const bf16* __restrict__ k,
                const float* __restrict__ maskv, float* __restrict__ att) {
  const long b = blockIdx.x;
  const int h = threadIdx.x >> 5;
  const int lane = threadIdx.x & 31, nl = lane & 15, hi = lane >> 4;
  const bf16* qb = q + b * (NM_ * 128) + h * 32;
  const bf16* kb = k + b * (NM_ * 128) + h * 32;
  const v16bf a  = frag_a(qb, 128);
  const v16bf bm = frag_b(kb, 128);
  v8f acc = {};
  acc = wmma_bf16(a, bm, acc);
  const float scale = 0.17677669529663689f;   // 1/sqrt(32)
#pragma unroll
  for (int i = 0; i < 8; ++i) {
    float x = acc[i] * scale;
    float mx = x;
#pragma unroll
    for (int m = 8; m >= 1; m >>= 1) mx = fmaxf(mx, __shfl_xor(mx, m, 16));
    const float e = __expf(x - mx);
    float s = e;
#pragma unroll
    for (int m = 8; m >= 1; m >>= 1) s += __shfl_xor(s, m, 16);
    const int row = i + hi * 8;
    const float val = e / s * maskv[b * 16 + row];
    att[((b * 4 + h) * 16 + row) * 16 + nl] = val;
  }
}

// ---------------- ctx = att@v, then ctx @ out_w[n] + residual mix -----------

__global__ __launch_bounds__(256)
void ctx_out_kernel(const bf16* __restrict__ vmat, const float* __restrict__ att,
                    const bf16* __restrict__ owt, const float* __restrict__ hb,
                    const float* __restrict__ hs, const float* __restrict__ maskv,
                    float* __restrict__ out) {
  __shared__ __align__(16) bf16 ctxs[16][1024];
  __shared__ __align__(16) bf16 Bs[256][40];
  __shared__ float atts[16][4][16];
  const int t = threadIdx.x;
  const int w = t >> 5, lane = t & 31, nl = lane & 15, hi = lane >> 4;
  const int n = blockIdx.y;
  const long b0 = (long)blockIdx.x * 16;
#pragma unroll
  for (int j = 0; j < 4; ++j) {
    const int flat = j * 256 + t;
    const int r = flat >> 6, rest = flat & 63;
    const int h = rest >> 4, m = rest & 15;
    atts[r][h][m] = att[(((b0 + r) * 4 + h) * 16 + n) * 16 + m];
  }
  __syncthreads();
  // ctx: batched 16-term matvec, 8 contiguous outputs per thread-chunk
#pragma unroll
  for (int j = 0; j < 8; ++j) {
    const int flat = j * 256 + t;                 // 2048 chunks of 8
    const int r = flat >> 7, c = (flat & 127) * 8;
    const int h = c >> 8;
    float a8[8];
#pragma unroll
    for (int i = 0; i < 8; ++i) a8[i] = 0.f;
#pragma unroll
    for (int m = 0; m < 16; ++m) {
      const uint4 pk = *(const uint4*)(vmat + ((b0 + r) * NM_ + m) * 1024 + c);
      const bf16* e = (const bf16*)&pk;
      const float am = atts[r][h][m];
#pragma unroll
      for (int i = 0; i < 8; ++i) a8[i] += am * (float)e[i];
    }
#pragma unroll
    for (int i = 0; i < 8; ++i) ctxs[r][c + i] = (bf16)a8[i];
  }
  __syncthreads();
  const bf16* Wn = owt + (long)n * 1024 * 256;    // [256 n][1024 k] bf16
  v8f acc[2];
  acc[0] = (v8f){};
  acc[1] = (v8f){};
  for (int k0 = 0; k0 < 1024; k0 += 32) {
#pragma unroll
    for (int j = 0; j < 4; ++j) {                 // Bt 256x32, async b128
      const int flat = j * 256 + t;
      const int c = flat >> 2, ch = (flat & 3) * 8;
      cp16(&Bs[c][ch], Wn + (long)c * 1024 + k0 + ch);
    }
    cp16_wait();
    __syncthreads();
    const v16bf a = frag_a(&ctxs[0][k0], 1024);
#pragma unroll
    for (int q2 = 0; q2 < 2; ++q2) {
      const v16bf b = frag_b(&Bs[w * 32 + q2 * 16][0], 40);
      acc[q2] = wmma_bf16(a, b, acc[q2]);
    }
    __syncthreads();
  }
#pragma unroll
  for (int q2 = 0; q2 < 2; ++q2) {
    const int col = w * 32 + q2 * 16 + nl;
#pragma unroll
    for (int i = 0; i < 8; ++i) {
      const int row = i + hi * 8;
      const long gi = ((b0 + row) * NM_ + n) * 256 + col;
      const float mk = maskv[(b0 + row) * NM_ + n];
      out[gi] = mk * (acc[q2][i] + hb[gi]) + (1.f - mk) * hs[gi];
    }
  }
}

// ---------------- Host-side orchestration -----------------------------------

extern "C" void kernel_launch(void* const* d_in, const int* in_sizes, int n_in,
                              void* d_out, int out_size, void* d_ws, size_t ws_size,
                              hipStream_t stream) {
  const float* x       = (const float*)d_in[0];
  const float* hs      = (const float*)d_in[1];
  const float* option  = (const float*)d_in[2];
  const float* value_w = (const float*)d_in[3];
  const float* value_b = (const float*)d_in[4];
  const float* p_w     = (const float*)d_in[5];
  const float* p_b     = (const float*)d_in[6];
  const float* x2h_w   = (const float*)d_in[7];
  const float* h2h_w   = (const float*)d_in[8];
  const float* q_w     = (const float*)d_in[9];
  const float* k_w     = (const float*)d_in[10];
  const float* v_w     = (const float*)d_in[11];
  const float* out_w   = (const float*)d_in[12];
  float* out = (float*)d_out;

  char* p = (char*)d_ws;
  auto alloc = [&](size_t bytes) -> char* {
    char* r = p;
    p += (bytes + 255) & ~(size_t)255;
    return r;
  };
  float* maskp0 = (float*)alloc((size_t)B_ * NM_ * 4);
  float* maskv  = (float*)alloc((size_t)B_ * NM_ * 4);
  float* xv     = (float*)alloc((size_t)B_ * VAL_ * 4);
  float* bias2  = (float*)alloc((size_t)NM_ * G3_ * 4);
  float* hb     = (float*)alloc((size_t)B_ * NM_ * HID_ * 4);
  bf16*  qb     = (bf16*)alloc((size_t)B_ * NM_ * 128 * 2);
  bf16*  kb     = (bf16*)alloc((size_t)B_ * NM_ * 128 * 2);
  bf16*  vb     = (bf16*)alloc((size_t)B_ * NM_ * 1024 * 2);
  float* att    = (float*)alloc((size_t)B_ * 4 * 16 * 16 * 4);
  // pre-transposed bf16 weights: [z][N][Kp]
  bf16* valwt = (bf16*)alloc((size_t)VAL_ * IN_ * 2);          // [400][512]
  bf16* x2ht  = (bf16*)alloc((size_t)NM_ * G3_ * VALP_ * 2);   // [16][768][416]
  bf16* h2ht  = (bf16*)alloc((size_t)NM_ * G3_ * HID_ * 2);    // [16][768][256]
  bf16* qwt   = (bf16*)alloc((size_t)NM_ * 128 * HID_ * 2);    // [16][128][256]
  bf16* kwt   = (bf16*)alloc((size_t)NM_ * 128 * HID_ * 2);
  bf16* vwt   = (bf16*)alloc((size_t)NM_ * 1024 * HID_ * 2);   // [16][1024][256]
  bf16* owt   = (bf16*)alloc((size_t)NM_ * 256 * 1024 * 2);    // [16][256][1024]

  // weight pre-pass (bf16 + transpose, K zero-padded to Kp)
  transpose_bf16_kernel<<<dim3(1, 13, 16), 256, 0, stream>>>(value_w, valwt, IN_, VAL_, IN_);
  transpose_bf16_kernel<<<dim3(NM_, 24, 13), 256, 0, stream>>>(x2h_w, x2ht, VAL_, G3_, VALP_);
  transpose_bf16_kernel<<<dim3(NM_, 24, 8), 256, 0, stream>>>(h2h_w, h2ht, HID_, G3_, HID_);
  transpose_bf16_kernel<<<dim3(NM_, 4, 8), 256, 0, stream>>>(q_w, qwt, HID_, 128, HID_);
  transpose_bf16_kernel<<<dim3(NM_, 4, 8), 256, 0, stream>>>(k_w, kwt, HID_, 128, HID_);
  transpose_bf16_kernel<<<dim3(NM_, 32, 8), 256, 0, stream>>>(v_w, vwt, HID_, 1024, HID_);
  transpose_bf16_kernel<<<dim3(NM_, 8, 32), 256, 0, stream>>>(out_w, owt, 1024, 256, 1024);

  bias2_kernel<<<dim3(NM_, 3, 1), 256, 0, stream>>>(value_b, x2h_w, bias2);
  gate_kernel<<<dim3(B_, 1, 1), 32, 0, stream>>>(option, p_w, p_b, maskp0, maskv);
  // xv = x @ value_w : (2048x512)x(512x400)
  gemm_kernel<false><<<dim3(B_ / 32, 7, 1), 256, 0, stream>>>(
      x, IN_, 0, valwt, 0, IN_, xv, VAL_, 0, VAL_, IN_, IN_);
  // fused gx/gh + GRU -> hb
  g_gru_kernel<<<dim3(B_ / 16, NM_, 1), 256, 0, stream>>>(
      xv, hs, x2ht, h2ht, bias2, maskp0, maskv, hb);
  // q,k : (2048x256)x(256x128) per n, bf16 out
  gemm_kernel<true><<<dim3(B_ / 32, 2, NM_), 256, 0, stream>>>(
      hb, NM_ * HID_, HID_, qwt, (long)128 * HID_, HID_, qb, (long)NM_ * 128, 128, 128, HID_, HID_);
  gemm_kernel<true><<<dim3(B_ / 32, 2, NM_), 256, 0, stream>>>(
      hb, NM_ * HID_, HID_, kwt, (long)128 * HID_, HID_, kb, (long)NM_ * 128, 128, 128, HID_, HID_);
  // v : (2048x256)x(256x1024) per n, bf16 out
  gemm_kernel<true><<<dim3(B_ / 32, 16, NM_), 256, 0, stream>>>(
      hb, NM_ * HID_, HID_, vwt, (long)1024 * HID_, HID_, vb, (long)NM_ * 1024, 1024, 1024, HID_, HID_);
  // attention scores + softmax + query mask
  att_kernel<<<dim3(B_, 1, 1), 128, 0, stream>>>(qb, kb, maskv, att);
  // ctx, out-projection, residual, final mask mix
  ctx_out_kernel<<<dim3(B_ / 16, NM_, 1), 256, 0, stream>>>(
      vb, att, owt, hb, hs, maskv, out);
}
